// Recognizer_36043365548708
// MI455X (gfx1250) — matrix operand order repaired and checked by reading, not compiled, not run on hardware
//
#include <hip/hip_runtime.h>
#include <cstdint>

// ---------------- problem constants (match reference) ----------------
#define B_    512
#define T_    128
#define C_    512
#define HL_   256          // per-direction LSTM hidden
#define H_    256          // decoder hidden
#define G_    1024         // 4*HL == 4*H
#define NCLS_ 97
#define S_    16           // MAXLEN+1 decode steps
#define CIN_  (C_ + NCLS_) // 609, row length of w_cell_ih

typedef __attribute__((ext_vector_type(16))) __bf16 v16bf;
typedef __attribute__((ext_vector_type(8)))  float  v8f;

// =====================================================================
// Packed fragment layouts (bf16), wave32 WMMA 16x16x32:
//   frag buffer = [tile][kchunk][lane(32)][16 bf16]   (32B per lane -> b128 loads)
// A (16x32 MxK): lane = m%16 + 16*hi, hi=(klocal>>3)&1, kk=klocal-8*hi,
//                elem e = kk<8 ? kk : kk-8
// B (32x16 KxN, B=W^T): lane = n%16 + 16*hi, hi=(klocal>>3)&1, elem e as for A
// =====================================================================

__device__ __forceinline__ v16bf fragld(const __bf16* __restrict__ P,
                                        long tile, int kc, int nkc) {
  return *(const v16bf*)(P + (((tile * nkc + kc) * 32) + (threadIdx.x & 31)) * 16);
}

// C/D (16x16 f32): VGPR v, lanes<16 -> M=v, lanes>=16 -> M=v+8; N = lane%16.
__device__ __forceinline__ void store_tile(float* __restrict__ out, long ldo,
                                           int m0, int n0, const v8f acc,
                                           const float* __restrict__ bias, int nmax) {
  const int lane = threadIdx.x & 31;
  const int n = n0 + (lane & 15);
  if (n >= nmax) return;
  const float bv = bias ? bias[n] : 0.0f;
  const int mb = m0 + ((lane & 16) ? 8 : 0);
#pragma unroll
  for (int v = 0; v < 8; ++v)
    out[(long)(mb + v) * ldo + n] = acc[v] + bv;
}

__device__ __forceinline__ v8f wmma_bf16(v16bf a, v16bf b, v8f c) {
  return __builtin_amdgcn_wmma_f32_16x16x32_bf16(false, a, false, b, (short)0, c, false, false);
}

__device__ __forceinline__ float sigmoidf_(float x) { return 1.0f / (1.0f + __expf(-x)); }

// packed A index from (m, k)
__device__ __forceinline__ long a_pack_idx(long m, int k, int nkc) {
  const int kc = k >> 5, kl = k & 31;
  const int hi = (kl >> 3) & 1;
  const int kk = kl - 8 * hi;
  const int e  = (kk < 8) ? kk : kk - 8;
  const int lane = (int)(m & 15) + 16 * hi;
  return (((m >> 4) * nkc + kc) * 32 + lane) * 16 + e;
}

// ---------------- pack kernels (run once per call) ----------------
// W [N x K] f32 row-major -> B-frag bf16 (rows n>=N zero-padded)
__global__ __launch_bounds__(256)
void pack_b_w(const float* __restrict__ W, __bf16* __restrict__ P,
              int N, int K, int ldw, long total) {
  const long gid = (long)blockIdx.x * 256 + threadIdx.x;
  if (gid >= total) return;
  const int e    = gid & 15;
  const int lane = (gid >> 4) & 31;
  const long rest = gid >> 9;
  const int nkc = K >> 5;
  const int kc  = (int)(rest % nkc);
  const int tn  = (int)(rest / nkc);
  const int n   = tn * 16 + (lane & 15);
  const int hi  = lane >> 4;
  const int k   = kc * 32 + ((e < 8) ? e : e + 8) + hi * 8;
  const float v = (n < N) ? W[(long)n * ldw + k] : 0.0f;
  P[gid] = (__bf16)v;
}

// feat_ct [B, T, C] -> per-step A frags: [t][tm(32)][kc(16)][lane][16]
__global__ __launch_bounds__(256)
void pack_a_featct(const float* __restrict__ feat_ct, __bf16* __restrict__ P) {
  const long gid = (long)blockIdx.x * 256 + threadIdx.x;   // B_*T_*C_ total
  const int e    = gid & 15;
  const int lane = (gid >> 4) & 31;
  const long rest = gid >> 9;
  const int kc = (int)(rest & 15);         // nkc = 16
  const int tm = (int)((rest >> 4) & 31);  // Mtiles = 32
  const int t  = (int)(rest >> 9);
  const int m  = tm * 16 + (lane & 15);
  const int hi = lane >> 4;
  const int klocal = ((e < 8) ? e : e + 8) + hi * 8;
  const int k = kc * 32 + klocal;
  P[gid] = (__bf16)feat_ct[((long)m * T_ + t) * C_ + k];
}

// ---------------- generic GEMM, 2x2 register blocking per wave ----------------
// out = A*W^T (+bias); Mtiles/Ntiles must be even; N-guarded store.
__global__ __launch_bounds__(256)
void gemm_pp_wmma(const __bf16* __restrict__ PA, const __bf16* __restrict__ PB,
                  const float* __restrict__ bias, float* __restrict__ out, long ldo,
                  int Msup, int Nsup, int N, int nkc) {
  const int wave = threadIdx.x >> 5;
  const int sup = blockIdx.x * (blockDim.x >> 5) + wave;
  if (sup >= Msup * Nsup) return;             // wave-uniform
  const long tm = 2L * (sup / Nsup);
  const long tn = 2L * (sup % Nsup);
  v8f a00 = {}, a01 = {}, a10 = {}, a11 = {};
  for (int kc = 0; kc < nkc; ++kc) {
    v16bf A0 = fragld(PA, tm,     kc, nkc);
    v16bf A1 = fragld(PA, tm + 1, kc, nkc);
    v16bf B0 = fragld(PB, tn,     kc, nkc);
    v16bf B1 = fragld(PB, tn + 1, kc, nkc);
    a00 = wmma_bf16(A0, B0, a00);
    a01 = wmma_bf16(A0, B1, a01);
    a10 = wmma_bf16(A1, B0, a10);
    a11 = wmma_bf16(A1, B1, a11);
  }
  store_tile(out, ldo, (int)tm * 16,       (int)tn * 16,        a00, bias, N);
  store_tile(out, ldo, (int)tm * 16,       (int)(tn + 1) * 16,  a01, bias, N);
  store_tile(out, ldo, (int)(tm + 1) * 16, (int)tn * 16,        a10, bias, N);
  store_tile(out, ldo, (int)(tm + 1) * 16, (int)(tn + 1) * 16,  a11, bias, N);
}

// ---------------- encoder: per-step gates = x_t*Wih^T + h*Whh^T (2x2) ----------------
__global__ __launch_bounds__(256)
void enc_step_wmma(const __bf16* __restrict__ pfeat_ct,
                   const __bf16* __restrict__ pwih_f, const __bf16* __restrict__ pwhh_f,
                   const __bf16* __restrict__ pwih_b, const __bf16* __restrict__ pwhh_b,
                   const __bf16* __restrict__ phf, const __bf16* __restrict__ phb,
                   float* __restrict__ gates_f, float* __restrict__ gates_b, int t) {
  const int dir = blockIdx.z;
  const int t_eff = dir ? (T_ - 1 - t) : t;
  const __bf16* pwih = dir ? pwih_b : pwih_f;
  const __bf16* pwhh = dir ? pwhh_b : pwhh_f;
  const __bf16* ph   = dir ? phb : phf;
  float* gates       = dir ? gates_b : gates_f;

  const int wave = threadIdx.x >> 5;
  const int sup = blockIdx.x * 8 + wave;      // 512 super: 16 (M) x 32 (N)
  const long tm = 2L * (sup >> 5);
  const long tn = 2L * (sup & 31);

  const __bf16* pa1 = pfeat_ct + (long)t_eff * (32L * 16 * 32 * 16); // per-t A frags
  v8f a00 = {}, a01 = {}, a10 = {}, a11 = {};
  for (int kc = 0; kc < 16; ++kc) {           // K = C_ = 512
    v16bf A0 = fragld(pa1, tm,     kc, 16);
    v16bf A1 = fragld(pa1, tm + 1, kc, 16);
    v16bf B0 = fragld(pwih, tn,     kc, 16);
    v16bf B1 = fragld(pwih, tn + 1, kc, 16);
    a00 = wmma_bf16(A0, B0, a00);
    a01 = wmma_bf16(A0, B1, a01);
    a10 = wmma_bf16(A1, B0, a10);
    a11 = wmma_bf16(A1, B1, a11);
  }
  for (int kc = 0; kc < 8; ++kc) {            // K = HL_ = 256
    v16bf A0 = fragld(ph, tm,     kc, 8);
    v16bf A1 = fragld(ph, tm + 1, kc, 8);
    v16bf B0 = fragld(pwhh, tn,     kc, 8);
    v16bf B1 = fragld(pwhh, tn + 1, kc, 8);
    a00 = wmma_bf16(A0, B0, a00);
    a01 = wmma_bf16(A0, B1, a01);
    a10 = wmma_bf16(A1, B0, a10);
    a11 = wmma_bf16(A1, B1, a11);
  }
  store_tile(gates, G_, (int)tm * 16,       (int)tn * 16,       a00, nullptr, G_);
  store_tile(gates, G_, (int)tm * 16,       (int)(tn + 1) * 16, a01, nullptr, G_);
  store_tile(gates, G_, (int)(tm + 1) * 16, (int)tn * 16,       a10, nullptr, G_);
  store_tile(gates, G_, (int)(tm + 1) * 16, (int)(tn + 1) * 16, a11, nullptr, G_);
}

__global__ __launch_bounds__(256)
void enc_cell(const float* __restrict__ gates_f, const float* __restrict__ gates_b,
              const float* __restrict__ b_f, const float* __restrict__ b_b,
              __bf16* __restrict__ phf, float* __restrict__ cf,
              __bf16* __restrict__ phb, float* __restrict__ cb,
              __bf16* __restrict__ pfeat, int t) {
  const int idx = blockIdx.x * blockDim.x + threadIdx.x;
  if (idx >= 2 * B_ * HL_) return;
  const int dir = idx / (B_ * HL_);
  const int r = idx % (B_ * HL_);
  const int b = r / HL_, j = r % HL_;
  const float* g    = (dir ? gates_b : gates_f) + (long)b * G_;
  const float* bias = dir ? b_b : b_f;
  const float gi  = g[j]            + bias[j];
  const float gfv = g[HL_ + j]      + bias[HL_ + j];
  const float gg  = g[2 * HL_ + j]  + bias[2 * HL_ + j];
  const float go  = g[3 * HL_ + j]  + bias[3 * HL_ + j];
  float* c = (dir ? cb : cf) + (long)b * HL_ + j;
  const float c2 = sigmoidf_(gfv) * (*c) + sigmoidf_(gi) * tanhf(gg);
  const float h2 = sigmoidf_(go) * tanhf(c2);
  *c = c2;
  __bf16* ph = dir ? phb : phf;
  ph[a_pack_idx(b, j, 8)] = (__bf16)h2;                    // h state, packed A [512x256]
  const int t_out = dir ? (T_ - 1 - t) : t;
  const long m = (long)b * T_ + t_out;
  pfeat[a_pack_idx(m, dir * HL_ + j, 16)] = (__bf16)h2;    // feat, packed A [65536x512]
}

// ---------------- decoder attention pieces ----------------
__global__ __launch_bounds__(256)
void attn_score(const float* __restrict__ feat_proj, const float* __restrict__ hp,
                const float* __restrict__ w_score, float* __restrict__ e) {
  const int wave = threadIdx.x >> 5, lane = threadIdx.x & 31;
  const int idx = blockIdx.x * 8 + wave;      // (b,t) pair
  if (idx >= B_ * T_) return;
  const int b = idx / T_;
  const float* fp  = feat_proj + (long)idx * H_;
  const float* hpb = hp + (long)b * H_;
  float s = 0.0f;
  for (int j = lane; j < H_; j += 32)
    s += tanhf(fp[j] + hpb[j]) * w_score[j];
#pragma unroll
  for (int o = 16; o > 0; o >>= 1) s += __shfl_xor(s, o, 32);
  if (lane == 0) e[idx] = s;
}

__global__ __launch_bounds__(128)
void softmax128(const float* __restrict__ e, float* __restrict__ alpha) {
  __shared__ float sm[128];
  const int b = blockIdx.x, t = threadIdx.x;
  const float v = e[b * T_ + t];
  sm[t] = v; __syncthreads();
  for (int o = 64; o > 0; o >>= 1) { if (t < o) sm[t] = fmaxf(sm[t], sm[t + o]); __syncthreads(); }
  const float mx = sm[0]; __syncthreads();
  const float ex = __expf(v - mx);
  sm[t] = ex; __syncthreads();
  for (int o = 64; o > 0; o >>= 1) { if (t < o) sm[t] += sm[t + o]; __syncthreads(); }
  alpha[b * T_ + t] = ex / sm[0];
}

// context[b,c] = sum_t alpha[b,t]*feat[b,t,c]; feat read from packed bf16,
// result written directly as packed A frag [512 x 512]
__global__ __launch_bounds__(256)
void context_k(const float* __restrict__ alpha, const __bf16* __restrict__ pfeat,
               __bf16* __restrict__ pctx) {
  const int idx = blockIdx.x * blockDim.x + threadIdx.x;
  if (idx >= B_ * C_) return;
  const int b = idx / C_, c = idx % C_;
  const float* al = alpha + (long)b * T_;
  const int kc = c >> 5, kl = c & 31;
  const int hi = (kl >> 3) & 1;
  const int kk = kl - 8 * hi;
  const int e  = (kk < 8) ? kk : kk - 8;
  float s = 0.0f;
  for (int t = 0; t < T_; ++t) {
    const long m = (long)b * T_ + t;
    const long pi = (((m >> 4) * 16 + kc) * 32 + (int)(m & 15) + 16 * hi) * 16 + e;
    s += al[t] * (float)pfeat[pi];
  }
  pctx[a_pack_idx(b, c, 16)] = (__bf16)s;
}

// per-subtile tail: bias + one-hot column gather, then store
__device__ __forceinline__ void dec_store(float* __restrict__ gates, int tm16, int tn16,
                                          const v8f acc, const float* __restrict__ b_cell,
                                          const float* __restrict__ w_cell_ih,
                                          const int* __restrict__ text, int s) {
  const int lane = threadIdx.x & 31;
  const int n  = tn16 + (lane & 15);
  const int mb = tm16 + ((lane & 16) ? 8 : 0);
  const float bv = b_cell[n];
  const float* wcol = w_cell_ih + (long)n * CIN_ + C_;
#pragma unroll
  for (int v = 0; v < 8; ++v) {
    const int m = mb + v;
    const int id = text[m * S_ + s];
    gates[(long)m * G_ + n] = acc[v] + bv + wcol[id];
  }
}

// gates = ctx*Wih[:, :512]^T + h*Whh^T + b_cell + Wih[:, 512+text[b,s]]  (2x2)
__global__ __launch_bounds__(256)
void dec_gates_wmma(const __bf16* __restrict__ pctx, const __bf16* __restrict__ phd,
                    const __bf16* __restrict__ pwcih, const __bf16* __restrict__ pwchh,
                    const float* __restrict__ w_cell_ih, const float* __restrict__ b_cell,
                    const int* __restrict__ text, float* __restrict__ gates, int s) {
  const int wave = threadIdx.x >> 5;
  const int sup = blockIdx.x * 8 + wave;      // 512 super: 16 x 32
  const long tm = 2L * (sup >> 5);
  const long tn = 2L * (sup & 31);
  v8f a00 = {}, a01 = {}, a10 = {}, a11 = {};
  for (int kc = 0; kc < 16; ++kc) {           // K = C_ = 512 (context part)
    v16bf A0 = fragld(pctx, tm,     kc, 16);
    v16bf A1 = fragld(pctx, tm + 1, kc, 16);
    v16bf B0 = fragld(pwcih, tn,     kc, 16);
    v16bf B1 = fragld(pwcih, tn + 1, kc, 16);
    a00 = wmma_bf16(A0, B0, a00);
    a01 = wmma_bf16(A0, B1, a01);
    a10 = wmma_bf16(A1, B0, a10);
    a11 = wmma_bf16(A1, B1, a11);
  }
  for (int kc = 0; kc < 8; ++kc) {            // K = H_ = 256 (hidden part)
    v16bf A0 = fragld(phd, tm,     kc, 8);
    v16bf A1 = fragld(phd, tm + 1, kc, 8);
    v16bf B0 = fragld(pwchh, tn,     kc, 8);
    v16bf B1 = fragld(pwchh, tn + 1, kc, 8);
    a00 = wmma_bf16(A0, B0, a00);
    a01 = wmma_bf16(A0, B1, a01);
    a10 = wmma_bf16(A1, B0, a10);
    a11 = wmma_bf16(A1, B1, a11);
  }
  dec_store(gates, (int)tm * 16,       (int)tn * 16,       a00, b_cell, w_cell_ih, text, s);
  dec_store(gates, (int)tm * 16,       (int)(tn + 1) * 16, a01, b_cell, w_cell_ih, text, s);
  dec_store(gates, (int)(tm + 1) * 16, (int)tn * 16,       a10, b_cell, w_cell_ih, text, s);
  dec_store(gates, (int)(tm + 1) * 16, (int)(tn + 1) * 16, a11, b_cell, w_cell_ih, text, s);
}

__global__ __launch_bounds__(256)
void dec_cell(const float* __restrict__ gates, __bf16* __restrict__ phd,
              float* __restrict__ c, __bf16* __restrict__ pouth, int s) {
  const int idx = blockIdx.x * blockDim.x + threadIdx.x;
  if (idx >= B_ * H_) return;
  const int b = idx / H_, j = idx % H_;
  const float* g = gates + (long)b * G_;
  const float gi = g[j], gfv = g[H_ + j], gg = g[2 * H_ + j], go = g[3 * H_ + j];
  const float c2 = sigmoidf_(gfv) * c[idx] + sigmoidf_(gi) * tanhf(gg);
  const float h2 = sigmoidf_(go) * tanhf(c2);
  c[idx] = c2;
  phd[a_pack_idx(b, j, 8)] = (__bf16)h2;                   // h_dec packed [512x256]
  pouth[a_pack_idx((long)b * S_ + s, j, 8)] = (__bf16)h2;  // out_h packed [8192x256]
}

// ---------------- host side ----------------
extern "C" void kernel_launch(void* const* d_in, const int* in_sizes, int n_in,
                              void* d_out, int out_size, void* d_ws, size_t ws_size,
                              hipStream_t stream) {
  const float* feat_ct   = (const float*)d_in[0];
  const int*   text      = (const int*)  d_in[1];
  // d_in[2] = max_len scalar (compile-time S_ = 16 used)
  const float* w_ih_f    = (const float*)d_in[3];
  const float* w_hh_f    = (const float*)d_in[4];
  const float* b_f       = (const float*)d_in[5];
  const float* w_ih_b    = (const float*)d_in[6];
  const float* w_hh_b    = (const float*)d_in[7];
  const float* b_b       = (const float*)d_in[8];
  const float* w_i2h     = (const float*)d_in[9];
  const float* w_h2h     = (const float*)d_in[10];
  const float* b_h2h     = (const float*)d_in[11];
  const float* w_score   = (const float*)d_in[12];
  const float* w_cell_ih = (const float*)d_in[13];
  const float* w_cell_hh = (const float*)d_in[14];
  const float* b_cell    = (const float*)d_in[15];
  const float* w_gen     = (const float*)d_in[16];
  const float* b_gen     = (const float*)d_in[17];
  float* out = (float*)d_out;

  // -------- workspace carve-out (byte offsets, all sizes 256B multiples) --------
  uint8_t* base = (uint8_t*)d_ws;
  size_t off = 0;
  auto takeF = [&](long n) { float*  p = (float*) (base + off); off += (size_t)n * 4; return p; };
  auto takeH = [&](long n) { __bf16* p = (__bf16*)(base + off); off += (size_t)n * 2; return p; };

  float* feat_proj = takeF(16777216L);           // [65536 x 256]
  float* gates_f   = takeF(524288L);             // [512 x 1024]
  float* gates_b   = takeF(524288L);
  float* hp        = takeF(131072L);             // [512 x 256]
  float* e         = takeF(65536L);              // [512 x 128]
  float* alpha     = takeF(65536L);
  float* gates_dec = takeF(524288L);
  // state region (single memset): cf, cb, c_dec (f32) + phf, phb, phdec (bf16)
  uint8_t* state0 = base + off;
  float*  cf    = takeF(131072L);
  float*  cb    = takeF(131072L);
  float*  c_dec = takeF(131072L);
  __bf16* phf   = takeH(131072L);
  __bf16* phb   = takeH(131072L);
  __bf16* phd   = takeH(131072L);
  size_t state_bytes = (size_t)((base + off) - state0);
  // packed bf16 operands
  __bf16* pfeat_ct = takeH(33554432L);           // per-t A frags
  __bf16* pfeat    = takeH(33554432L);           // A [65536 x 512]
  __bf16* pctx     = takeH(262144L);             // A [512 x 512]
  __bf16* pouth    = takeH(2097152L);            // A [8192 x 256]
  __bf16* pwih_f   = takeH(524288L);             // B [1024 x 512]
  __bf16* pwih_b   = takeH(524288L);
  __bf16* pwhh_f   = takeH(262144L);             // B [1024 x 256]
  __bf16* pwhh_b   = takeH(262144L);
  __bf16* pwi2h    = takeH(131072L);             // B [256 x 512]
  __bf16* pwh2h    = takeH(65536L);              // B [256 x 256]
  __bf16* pwcih    = takeH(524288L);             // B [1024 x 512] (cols 0..511)
  __bf16* pwchh    = takeH(262144L);             // B [1024 x 256]
  __bf16* pwgen    = takeH(32768L);              // B [128 x 256] zero-padded (8 tiles)

  const dim3 blk256(256), blk128(128);

  // zero LSTM states
  hipMemsetAsync(state0, 0, state_bytes, stream);

  // -------- one-time packs (weights + feat_ct) --------
  auto packB = [&](const float* W, __bf16* P, int N, int K, int ldw, int Ntiles) {
    const long total = (long)Ntiles * 16 * K;
    pack_b_w<<<dim3((unsigned)((total + 255) / 256)), blk256, 0, stream>>>(W, P, N, K, ldw, total);
  };
  packB(w_ih_f,    pwih_f, 1024, 512, 512, 64);
  packB(w_ih_b,    pwih_b, 1024, 512, 512, 64);
  packB(w_hh_f,    pwhh_f, 1024, 256, 256, 64);
  packB(w_hh_b,    pwhh_b, 1024, 256, 256, 64);
  packB(w_i2h,     pwi2h,   256, 512, 512, 16);
  packB(w_h2h,     pwh2h,   256, 256, 256, 16);
  packB(w_cell_ih, pwcih,  1024, 512, CIN_, 64);   // first 512 columns (context part)
  packB(w_cell_hh, pwchh,  1024, 256, 256, 64);
  packB(w_gen,     pwgen,  NCLS_, 256, 256, 8);    // padded to 8 N-tiles
  pack_a_featct<<<dim3(131072), blk256, 0, stream>>>(feat_ct, pfeat_ct);

  // -------- BiLSTM encoder: 128 sequential steps, both directions per launch --------
  for (int t = 0; t < T_; ++t) {
    enc_step_wmma<<<dim3(64, 1, 2), blk256, 0, stream>>>(
        pfeat_ct, pwih_f, pwhh_f, pwih_b, pwhh_b, phf, phb, gates_f, gates_b, t);
    enc_cell<<<dim3((2 * B_ * HL_) / 256), blk256, 0, stream>>>(
        gates_f, gates_b, b_f, b_b, phf, cf, phb, cb, pfeat, t);
  }

  // -------- step-invariant projection: feat_proj = feat @ w_i2h^T --------
  gemm_pp_wmma<<<dim3((2048 * 8) / 8), blk256, 0, stream>>>(
      pfeat, pwi2h, nullptr, feat_proj, H_, 2048, 8, H_, 16);

  // -------- attention decoder: 16 sequential steps --------
  for (int s = 0; s < S_; ++s) {
    gemm_pp_wmma<<<dim3((16 * 8) / 8), blk256, 0, stream>>>(
        phd, pwh2h, b_h2h, hp, H_, 16, 8, H_, 8);
    attn_score<<<dim3((B_ * T_) / 8), blk256, 0, stream>>>(feat_proj, hp, w_score, e);
    softmax128<<<dim3(B_), blk128, 0, stream>>>(e, alpha);
    context_k<<<dim3((B_ * C_) / 256), blk256, 0, stream>>>(alpha, pfeat, pctx);
    dec_gates_wmma<<<dim3(64), blk256, 0, stream>>>(
        pctx, phd, pwcih, pwchh, w_cell_ih, b_cell, text, gates_dec, s);
    dec_cell<<<dim3((B_ * H_) / 256), blk256, 0, stream>>>(gates_dec, phd, c_dec, pouth, s);
  }

  // -------- generator: probs = out_h @ w_gen^T + b_gen  [8192 x 97] --------
  gemm_pp_wmma<<<dim3((256 * 4) / 8), blk256, 0, stream>>>(
      pouth, pwgen, b_gen, out, NCLS_, 256, 4, NCLS_, 8);
}
